// AdaDiMT_27917287424467
// MI455X (gfx1250) — compile-verified, matrix operand order checked by reading, not run on hardware
//
#include <hip/hip_runtime.h>

// ---------------- problem constants ----------------
#define B_   4
#define L_   2048
#define H_   512
#define DI_  1024
#define DS_  16
#define DC_  4
#define DTR_ 32
#define MH_  2048
#define M_   (B_ * L_)     // 8192 token rows

typedef unsigned short us;
typedef __attribute__((ext_vector_type(16))) __bf16          v16bf;
typedef __attribute__((ext_vector_type(8)))  unsigned short  us8;
typedef __attribute__((ext_vector_type(8)))  float           v8f;

// ---------------- helpers ----------------
__device__ __forceinline__ us f2bf(float f) {
  unsigned int u = __float_as_uint(f);
  unsigned int r = u + 0x7FFFu + ((u >> 16) & 1u);   // round-to-nearest-even
  return (us)(r >> 16);
}
__device__ __forceinline__ float bf2f(us h) {
  return __uint_as_float(((unsigned int)h) << 16);
}
__device__ __forceinline__ float silu_f(float x) { return x / (1.0f + __expf(-x)); }

// CDNA5 async global->LDS copy (16 bytes per lane), tracked by ASYNCcnt.
// GV addressing: dst operand is a VGPR holding the LDS byte address
// (= low 32 bits of the generic shared pointer, per the flat LDS aperture rule).
__device__ __forceinline__ void async_copy16(const us* gsrc, us* ldst) {
  unsigned lds = (unsigned)(unsigned long long)ldst;
  asm volatile("global_load_async_to_lds_b128 %0, %1, off"
               :: "v"(lds), "v"(gsrc) : "memory");
}
// Async loads complete in order -> waiting for <=N leaves only the N most
// recently issued transfers (the next tile) outstanding.
#define WAIT_ASYNC(nrem) asm volatile("s_wait_asynccnt " #nrem ::: "memory")

union Frag { v16bf v; us8 h[2]; };

// Build a 16x32 bf16 WMMA operand fragment from an LDS tile (row-major, ld=32).
// ISA layout: lane l holds row (l&15); K 0..7,16..23 in lanes<16 / 8..15,24..31 in lanes>=16,
// each VGPR holds a K-pair -> two contiguous 8-halfword chunks per lane.
__device__ __forceinline__ v16bf load_frag(const us* base, int row0, int lane) {
  Frag f;
  int r  = row0 + (lane & 15);
  int k0 = (lane >> 4) << 3;
  f.h[0] = *(const us8*)(base + r * 32 + k0);
  f.h[1] = *(const us8*)(base + r * 32 + k0 + 16);
  return f.v;
}

// ---------------- generic bf16 GEMM: C[M,N] = A[M,K] @ W[N,K]^T ----------------
// Block tile 128x64, 8 waves, each wave 32x32 (2x2 WMMA f32_16x16x32_bf16 tiles).
// Tiles stream into double-buffered LDS with GLOBAL_LOAD_ASYNC_TO_LDS_B128:
// tile k+1 is issued before waiting on tile k, hiding global latency behind WMMA.
// EPI: 0 = plain f32 store, 1 = softplus(acc+bias), 2 = resid + gate*acc,
//      3 = resid + gate*(acc+bias)
template <int EPI>
__global__ __launch_bounds__(256) void gemm_bf16_kernel(
    const us* __restrict__ A, long lda,
    const us* __restrict__ Bw,           // (N,K) row-major
    int N, int K,
    float* __restrict__ C,
    const float* __restrict__ resid,
    const float* __restrict__ ada,       // (B, 3072)
    const float* __restrict__ bias,
    int gate_off)
{
  __shared__ us As[2][128 * 32];
  __shared__ us Bs[2][64 * 32];

  const int tid  = threadIdx.x;
  const int lane = tid & 31;
  const int wid  = tid >> 5;
  const int rowBase = blockIdx.y * 128;
  const int colBase = blockIdx.x * 64;
  const int wm = (wid >> 1) * 32;
  const int wn = (wid & 1) * 32;

  // per-thread staging coordinates (3 x 16B async transfers per tile)
  const int eA0 = tid * 16, eA1 = tid * 16 + 8, eB = tid * 8;
  const long aOff0 = (long)(rowBase + (eA0 >> 5)) * lda + (eA0 & 31);
  const long aOff1 = (long)(rowBase + (eA1 >> 5)) * lda + (eA1 & 31);
  const long bOff  = (long)(colBase + (eB  >> 5)) * K   + (eB  & 31);

  auto stage = [&](int k0, int buf) {
    async_copy16(A  + aOff0 + k0, &As[buf][eA0]);
    async_copy16(A  + aOff1 + k0, &As[buf][eA1]);
    async_copy16(Bw + bOff  + k0, &Bs[buf][eB]);
  };

  v8f acc[2][2] = {};

  stage(0, 0);
  int buf = 0;
  for (int k0 = 0; k0 < K; k0 += 32, buf ^= 1) {
    if (k0 + 32 < K) {
      stage(k0 + 32, buf ^ 1);      // stream next tile behind this tile's math
      WAIT_ASYNC(0x3);              // current tile landed; next (3) in flight
    } else {
      WAIT_ASYNC(0x0);
    }
    if (k0 + 64 < K) {              // L2 prefetch two tiles ahead
      __builtin_prefetch(A + aOff0 + k0 + 64, 0, 1);
      __builtin_prefetch(Bw + bOff + k0 + 64, 0, 1);
    }
    __syncthreads();                // tile visible workgroup-wide

    v16bf a0 = load_frag(As[buf], wm,      lane);
    v16bf a1 = load_frag(As[buf], wm + 16, lane);
    v16bf b0 = load_frag(Bs[buf], wn,      lane);
    v16bf b1 = load_frag(Bs[buf], wn + 16, lane);
    acc[0][0] = __builtin_amdgcn_wmma_f32_16x16x32_bf16(false, a0, false, b0, (short)0, acc[0][0], false, false);
    acc[0][1] = __builtin_amdgcn_wmma_f32_16x16x32_bf16(false, a0, false, b1, (short)0, acc[0][1], false, false);
    acc[1][0] = __builtin_amdgcn_wmma_f32_16x16x32_bf16(false, a1, false, b0, (short)0, acc[1][0], false, false);
    acc[1][1] = __builtin_amdgcn_wmma_f32_16x16x32_bf16(false, a1, false, b1, (short)0, acc[1][1], false, false);
    __syncthreads();                // all waves done reading buf -> reusable
  }

  // C/D layout: VGPR e -> row e (+8 for lanes>=16), col = lane&15
#pragma unroll
  for (int i = 0; i < 2; ++i) {
#pragma unroll
    for (int j = 0; j < 2; ++j) {
#pragma unroll
      for (int e = 0; e < 8; ++e) {
        int r = rowBase + wm + i * 16 + ((lane >> 4) << 3) + e;
        int cc = colBase + wn + j * 16 + (lane & 15);
        float v = acc[i][j][e];
        long idx = (long)r * N + cc;
        if constexpr (EPI == 0) {
          C[idx] = v;
        } else if constexpr (EPI == 1) {
          v += bias[cc];
          C[idx] = (v > 20.f) ? v : log1pf(__expf(v));
        } else if constexpr (EPI == 2) {
          int b = r / L_;
          C[idx] = resid[idx] + ada[b * 3072 + gate_off + cc] * v;
        } else {
          int b = r / L_;
          C[idx] = resid[idx] + ada[b * 3072 + gate_off + cc] * (v + bias[cc]);
        }
      }
    }
  }
}

// ---------------- fc1 + GLU fused (dual column tiles: n and n+MH) ----------------
__global__ __launch_bounds__(256) void fc1_glu_kernel(
    const us* __restrict__ A,          // (M, H) bf16
    const us* __restrict__ Bw,         // (2*MH, H) bf16
    const float* __restrict__ bias,    // (2*MH,)
    us* __restrict__ Out)              // (M, MH) bf16 = u * gelu(z)
{
  __shared__ us As[2][128 * 32];
  __shared__ us Bs[2][2][64 * 32];

  const int tid  = threadIdx.x;
  const int lane = tid & 31;
  const int wid  = tid >> 5;
  const int rowBase = blockIdx.y * 128;
  const int colBase = blockIdx.x * 64;
  const int wm = (wid >> 1) * 32;
  const int wn = (wid & 1) * 32;

  const int eA0 = tid * 16, eA1 = tid * 16 + 8, eB = tid * 8;
  const long aOff0 = (long)(rowBase + (eA0 >> 5)) * H_ + (eA0 & 31);
  const long aOff1 = (long)(rowBase + (eA1 >> 5)) * H_ + (eA1 & 31);
  const long bOffU = (long)(colBase + (eB >> 5)) * H_ + (eB & 31);
  const long bOffZ = (long)(colBase + MH_ + (eB >> 5)) * H_ + (eB & 31);

  auto stage = [&](int k0, int buf) {
    async_copy16(A  + aOff0 + k0, &As[buf][eA0]);
    async_copy16(A  + aOff1 + k0, &As[buf][eA1]);
    async_copy16(Bw + bOffU + k0, &Bs[buf][0][eB]);
    async_copy16(Bw + bOffZ + k0, &Bs[buf][1][eB]);
  };

  v8f accU[2][2] = {}, accZ[2][2] = {};

  stage(0, 0);
  int buf = 0;
  for (int k0 = 0; k0 < H_; k0 += 32, buf ^= 1) {
    if (k0 + 32 < H_) {
      stage(k0 + 32, buf ^ 1);
      WAIT_ASYNC(0x4);              // current tile landed; next (4) in flight
    } else {
      WAIT_ASYNC(0x0);
    }
    __syncthreads();

    v16bf a0  = load_frag(As[buf], wm,      lane);
    v16bf a1  = load_frag(As[buf], wm + 16, lane);
    v16bf bu0 = load_frag(Bs[buf][0], wn,      lane);
    v16bf bu1 = load_frag(Bs[buf][0], wn + 16, lane);
    v16bf bz0 = load_frag(Bs[buf][1], wn,      lane);
    v16bf bz1 = load_frag(Bs[buf][1], wn + 16, lane);
    accU[0][0] = __builtin_amdgcn_wmma_f32_16x16x32_bf16(false, a0, false, bu0, (short)0, accU[0][0], false, false);
    accU[0][1] = __builtin_amdgcn_wmma_f32_16x16x32_bf16(false, a0, false, bu1, (short)0, accU[0][1], false, false);
    accU[1][0] = __builtin_amdgcn_wmma_f32_16x16x32_bf16(false, a1, false, bu0, (short)0, accU[1][0], false, false);
    accU[1][1] = __builtin_amdgcn_wmma_f32_16x16x32_bf16(false, a1, false, bu1, (short)0, accU[1][1], false, false);
    accZ[0][0] = __builtin_amdgcn_wmma_f32_16x16x32_bf16(false, a0, false, bz0, (short)0, accZ[0][0], false, false);
    accZ[0][1] = __builtin_amdgcn_wmma_f32_16x16x32_bf16(false, a0, false, bz1, (short)0, accZ[0][1], false, false);
    accZ[1][0] = __builtin_amdgcn_wmma_f32_16x16x32_bf16(false, a1, false, bz0, (short)0, accZ[1][0], false, false);
    accZ[1][1] = __builtin_amdgcn_wmma_f32_16x16x32_bf16(false, a1, false, bz1, (short)0, accZ[1][1], false, false);
    __syncthreads();
  }

#pragma unroll
  for (int i = 0; i < 2; ++i) {
#pragma unroll
    for (int j = 0; j < 2; ++j) {
#pragma unroll
      for (int e = 0; e < 8; ++e) {
        int r  = rowBase + wm + i * 16 + ((lane >> 4) << 3) + e;
        int cc = colBase + wn + j * 16 + (lane & 15);
        float u = accU[i][j][e] + bias[cc];
        float z = accZ[i][j][e] + bias[cc + MH_];
        float g = 0.5f * z * (1.0f + tanhf(0.7978845608f * (z + 0.044715f * z * z * z)));
        Out[(long)r * MH_ + cc] = f2bf(u * g);
      }
    }
  }
}

// ---------------- adaLN: ada = silu(c) @ ada_w^T + ada_b ----------------
__global__ void ada_kernel(const float* __restrict__ c, const float* __restrict__ w,
                           const float* __restrict__ b, float* __restrict__ ada)
{
  int o = blockIdx.x * blockDim.x + threadIdx.x;
  if (o >= B_ * 3072) return;
  int bb = o / 3072, j = o % 3072;
  float s = 0.f;
  for (int k = 0; k < H_; ++k) s += silu_f(c[bb * H_ + k]) * w[j * H_ + k];
  ada[o] = s + b[j];
}

// ---------------- rmsnorm + modulate -> bf16 ----------------
__global__ __launch_bounds__(256) void norm_mod_kernel(
    const float* __restrict__ x, const float* __restrict__ w,
    const float* __restrict__ ada, int sh_off, int sc_off,
    us* __restrict__ out)
{
  __shared__ float red[256];
  long r = blockIdx.x;
  int  b = (int)(r / L_);
  const float* xr = x + r * H_;
  int c0 = threadIdx.x, c1 = threadIdx.x + 256;
  float v0 = xr[c0], v1 = xr[c1];
  red[threadIdx.x] = v0 * v0 + v1 * v1;
  __syncthreads();
  for (int s = 128; s > 0; s >>= 1) {
    if (threadIdx.x < s) red[threadIdx.x] += red[threadIdx.x + s];
    __syncthreads();
  }
  float rs = rsqrtf(red[0] / (float)H_ + 1e-5f);
  const float* ab = ada + b * 3072;
  out[r * H_ + c0] = f2bf(v0 * rs * w[c0] * (1.f + ab[sc_off + c0]) + ab[sh_off + c0]);
  out[r * H_ + c1] = f2bf(v1 * rs * w[c1] * (1.f + ab[sc_off + c1]) + ab[sh_off + c1]);
}

// ---------------- depthwise causal conv + silu (optionally on reversed time) ----------------
__global__ void conv_silu_kernel(const float* __restrict__ xz, const float* __restrict__ w,
                                 const float* __restrict__ bias, us* __restrict__ xc, int rev)
{
  long gid = (long)blockIdx.x * blockDim.x + threadIdx.x;
  if (gid >= (long)M_ * DI_) return;
  int  d   = (int)(gid % DI_);
  long row = gid / DI_;
  int  t   = (int)(row % L_);
  int  b   = (int)(row / L_);
  float acc = bias[d];
#pragma unroll
  for (int k = 0; k < DC_; ++k) {
    int st = t - (DC_ - 1) + k;
    if (st >= 0) {
      int tt = rev ? (L_ - 1 - st) : st;
      acc += w[d * DC_ + k] * xz[(long)(b * L_ + tt) * (2 * DI_) + d];
    }
  }
  xc[gid] = f2bf(silu_f(acc));
}

// ---------------- slice dbl[:, :32] -> bf16 (dtproj A operand) ----------------
__global__ void slice_cvt_kernel(const float* __restrict__ dbl, us* __restrict__ dtr)
{
  long gid = (long)blockIdx.x * blockDim.x + threadIdx.x;
  if (gid >= (long)M_ * DTR_) return;
  long row = gid / DTR_;
  int  cc  = (int)(gid % DTR_);
  dtr[gid] = f2bf(dbl[row * 64 + cc]);
}

// ---------------- selective scan (sequential over L, lane = channel) ----------------
__global__ __launch_bounds__(256) void scan_kernel(
    const us* __restrict__ xc,    // (B*L, DI) bf16 (scan-time order)
    const float* __restrict__ dt, // (B*L, DI)
    const float* __restrict__ dbl,// (B*L, 64): cols [32,48)=Bm, [48,64)=Cm
    const float* __restrict__ A_log, const float* __restrict__ Dp,
    float* __restrict__ y)        // (B*L, DI)
{
  __shared__ float bc[32];
  int d = blockIdx.x * 256 + threadIdx.x;
  int b = blockIdx.y;
  float Arow[DS_], h[DS_];
#pragma unroll
  for (int n = 0; n < DS_; ++n) { Arow[n] = -__expf(A_log[d * DS_ + n]); h[n] = 0.f; }
  float Dd = Dp[d];
  for (int t = 0; t < L_; ++t) {
    long row = (long)b * L_ + t;
    if (threadIdx.x < 32) bc[threadIdx.x] = dbl[row * 64 + 32 + threadIdx.x];
    __syncthreads();
    float u = bf2f(xc[row * DI_ + d]);
    float delta = dt[row * DI_ + d];
    float du = delta * u;
    float yv = 0.f;
#pragma unroll
    for (int n = 0; n < DS_; ++n) {
      float dA = __expf(delta * Arow[n]);
      h[n] = dA * h[n] + du * bc[n];
      yv += h[n] * bc[DS_ + n];
    }
    y[row * DI_ + d] = yv + Dd * u;
    __syncthreads();
  }
}

// ---------------- combine directions + silu(z) gate -> bf16 ----------------
__global__ void gate_kernel(const float* __restrict__ yf, const float* __restrict__ yb,
                            const float* __restrict__ xz, us* __restrict__ gated)
{
  long gid = (long)blockIdx.x * blockDim.x + threadIdx.x;
  if (gid >= (long)M_ * DI_) return;
  int  d   = (int)(gid % DI_);
  long row = gid / DI_;
  int  t   = (int)(row % L_);
  int  b   = (int)(row / L_);
  long rrow = (long)b * L_ + (L_ - 1 - t);
  float z = xz[row * (2 * DI_) + DI_ + d];
  gated[gid] = f2bf((yf[gid] + yb[rrow * DI_ + d]) * silu_f(z));
}

// ---------------- f32 -> bf16 weight conversion ----------------
__global__ void convert_kernel(const float* __restrict__ s, us* __restrict__ d, long n)
{
  long i = (long)blockIdx.x * blockDim.x + threadIdx.x;
  if (i < n) d[i] = f2bf(s[i]);
}

// ---------------- host orchestration ----------------
extern "C" void kernel_launch(void* const* d_in, const int* in_sizes, int n_in,
                              void* d_out, int out_size, void* d_ws, size_t ws_size,
                              hipStream_t stream)
{
  (void)in_sizes; (void)n_in; (void)out_size; (void)ws_size;
  const float* x          = (const float*)d_in[0];
  const float* c          = (const float*)d_in[1];
  const float* rms1_w     = (const float*)d_in[2];
  const float* rms2_w     = (const float*)d_in[3];
  const float* in_proj_w  = (const float*)d_in[4];
  const float* out_proj_w = (const float*)d_in[5];
  const float* ada_w      = (const float*)d_in[6];
  const float* ada_bias   = (const float*)d_in[7];
  const float* fc1_w      = (const float*)d_in[8];
  const float* fc1_b      = (const float*)d_in[9];
  const float* fc2_w      = (const float*)d_in[10];
  const float* fc2_b      = (const float*)d_in[11];
  const float* conv_w[2]   = {(const float*)d_in[12], (const float*)d_in[19]};
  const float* conv_b[2]   = {(const float*)d_in[13], (const float*)d_in[20]};
  const float* xproj_w[2]  = {(const float*)d_in[14], (const float*)d_in[21]};
  const float* dtproj_w[2] = {(const float*)d_in[15], (const float*)d_in[22]};
  const float* dtproj_b[2] = {(const float*)d_in[16], (const float*)d_in[23]};
  const float* A_log[2]    = {(const float*)d_in[17], (const float*)d_in[24]};
  const float* Dp[2]       = {(const float*)d_in[18], (const float*)d_in[25]};
  float* out = (float*)d_out;

  char* p = (char*)d_ws;
  auto alloc = [&p](size_t bytes) -> void* {
    void* q = (void*)p;
    p += (bytes + 255) & ~((size_t)255);
    return q;
  };

  us* wb_in  = (us*)alloc((size_t)2 * DI_ * H_ * 2);
  us* wb_out = (us*)alloc((size_t)H_ * DI_ * 2);
  us* wb_fc1 = (us*)alloc((size_t)2 * MH_ * H_ * 2);
  us* wb_fc2 = (us*)alloc((size_t)H_ * MH_ * 2);
  us* wb_xp[2];  wb_xp[0]  = (us*)alloc((size_t)64 * DI_ * 2);  wb_xp[1]  = (us*)alloc((size_t)64 * DI_ * 2);
  us* wb_dtp[2]; wb_dtp[0] = (us*)alloc((size_t)DI_ * DTR_ * 2); wb_dtp[1] = (us*)alloc((size_t)DI_ * DTR_ * 2);
  float* adab = (float*)alloc((size_t)B_ * 3072 * 4);
  us*    t1   = (us*)alloc((size_t)M_ * H_ * 2);           // reused for norm2 output
  float* xz   = (float*)alloc((size_t)M_ * 2 * DI_ * 4);
  us* xcb[2]; float* dblb[2]; us* dtrb[2]; float* dtb[2]; float* ybuf[2];
  for (int dir = 0; dir < 2; ++dir) {
    xcb[dir]  = (us*)alloc((size_t)M_ * DI_ * 2);
    dblb[dir] = (float*)alloc((size_t)M_ * 64 * 4);
    dtrb[dir] = (us*)alloc((size_t)M_ * DTR_ * 2);
    dtb[dir]  = (float*)alloc((size_t)M_ * DI_ * 4);
    ybuf[dir] = (float*)alloc((size_t)M_ * DI_ * 4);
  }
  float* x2    = (float*)alloc((size_t)M_ * H_ * 4);
  us*    gated = (us*)alloc((size_t)M_ * DI_ * 2);
  us*    mlp   = (us*)alloc((size_t)M_ * MH_ * 2);

  auto cvt = [&](const float* s, us* dPtr, long n) {
    convert_kernel<<<(unsigned)((n + 255) / 256), 256, 0, stream>>>(s, dPtr, n);
  };
  cvt(in_proj_w,  wb_in,  2L * DI_ * H_);
  cvt(out_proj_w, wb_out, (long)H_ * DI_);
  cvt(fc1_w,      wb_fc1, 2L * MH_ * H_);
  cvt(fc2_w,      wb_fc2, (long)H_ * MH_);
  for (int dir = 0; dir < 2; ++dir) {
    cvt(xproj_w[dir],  wb_xp[dir],  64L * DI_);
    cvt(dtproj_w[dir], wb_dtp[dir], (long)DI_ * DTR_);
  }

  // adaLN vector
  ada_kernel<<<(B_ * 3072 + 255) / 256, 256, 0, stream>>>(c, ada_w, ada_bias, adab);

  // norm1 + modulate -> t1 (bf16), then xz = t1 @ in_proj^T
  norm_mod_kernel<<<M_, 256, 0, stream>>>(x, rms1_w, adab, 0, 512, t1);
  gemm_bf16_kernel<0><<<dim3((2 * DI_) / 64, M_ / 128), 256, 0, stream>>>(
      t1, H_, wb_in, 2 * DI_, H_, xz, nullptr, nullptr, nullptr, 0);

  // per-direction Mamba branch (backward runs on reversed-time buffers)
  for (int dir = 0; dir < 2; ++dir) {
    conv_silu_kernel<<<(unsigned)(((long)M_ * DI_ + 255) / 256), 256, 0, stream>>>(
        xz, conv_w[dir], conv_b[dir], xcb[dir], dir);
    gemm_bf16_kernel<0><<<dim3(1, M_ / 128), 256, 0, stream>>>(
        xcb[dir], DI_, wb_xp[dir], 64, DI_, dblb[dir], nullptr, nullptr, nullptr, 0);
    slice_cvt_kernel<<<(unsigned)(((long)M_ * DTR_ + 255) / 256), 256, 0, stream>>>(
        dblb[dir], dtrb[dir]);
    gemm_bf16_kernel<1><<<dim3(DI_ / 64, M_ / 128), 256, 0, stream>>>(
        dtrb[dir], DTR_, wb_dtp[dir], DI_, DTR_, dtb[dir], nullptr, nullptr, dtproj_b[dir], 0);
    scan_kernel<<<dim3(DI_ / 256, B_), 256, 0, stream>>>(
        xcb[dir], dtb[dir], dblb[dir], A_log[dir], Dp[dir], ybuf[dir]);
  }

  // combine + gate, out_proj with residual + g_m
  gate_kernel<<<(unsigned)(((long)M_ * DI_ + 255) / 256), 256, 0, stream>>>(
      ybuf[0], ybuf[1], xz, gated);
  gemm_bf16_kernel<2><<<dim3(H_ / 64, M_ / 128), 256, 0, stream>>>(
      gated, DI_, wb_out, H_, DI_, x2, x, adab, nullptr, 1024);

  // norm2 + modulate -> t1, fc1+GLU fused, fc2 with residual + g_p -> out
  norm_mod_kernel<<<M_, 256, 0, stream>>>(x2, rms2_w, adab, 1536, 2048, t1);
  fc1_glu_kernel<<<dim3(MH_ / 64, M_ / 128), 256, 0, stream>>>(t1, wb_fc1, fc1_b, mlp);
  gemm_bf16_kernel<3><<<dim3(H_ / 64, M_ / 128), 256, 0, stream>>>(
      mlp, MH_, wb_fc2, H_, MH_, out, x2, adab, fc2_b, 2560);
}
